// AttentionModule_22110491639946
// MI455X (gfx1250) — matrix operand order repaired and checked by reading, not compiled, not run on hardware
//
#include <hip/hip_runtime.h>

// Problem constants (from reference)
#define S_    512
#define AD_   768
#define BD_   768
#define H_    12
#define B_    16
#define DH_   64
#define MROWS (B_ * S_)                 // 8192
#define INV_SCALE 0.03608439182435161f  // 1/sqrt(768)

typedef __attribute__((ext_vector_type(16))) __bf16 bf16x16;
typedef __attribute__((ext_vector_type(8)))  __bf16 bf16x8;
typedef __attribute__((ext_vector_type(8)))  float  f32x8;

static __device__ __forceinline__ unsigned short cvt_bf16(float f) {
    unsigned int u = __float_as_uint(f);
    unsigned int r = 0x7FFFu + ((u >> 16) & 1u);   // round-to-nearest-even
    return (unsigned short)((u + r) >> 16);
}

// Load a 16-element bf16 fragment as two contiguous 8-element (16B) LDS reads.
// s0/s1 are element offsets within the row (multiples of 8 -> 16B aligned).
static __device__ __forceinline__ bf16x16 frag_ld(const unsigned short* rowp, int s0, int s1) {
    bf16x8 lo = *(const bf16x8*)(rowp + s0);
    bf16x8 hi = *(const bf16x8*)(rowp + s1);
    return __builtin_shufflevector(lo, hi, 0,1,2,3,4,5,6,7,8,9,10,11,12,13,14,15);
}

static __device__ __forceinline__ f32x8 wmma_bf16(bf16x16 a, bf16x16 b, f32x8 c) {
    return __builtin_amdgcn_wmma_f32_16x16x32_bf16(false, a, false, b, (short)0, c,
                                                   false, false);
}

// Convert one float4 (16 floats total would be 4 of these) into 4 bf16 LDS slots.
static __device__ __forceinline__ void cvt4_store(unsigned short* d, float4 x) {
    d[0] = cvt_bf16(x.x); d[1] = cvt_bf16(x.y); d[2] = cvt_bf16(x.z); d[3] = cvt_bf16(x.w);
}

// -----------------------------------------------------------------------------
// Y[M=8192, 768] = X[8192, 768] @ W[768, 768]^T (+ bias).  W is [out,in], so
// both A (=X) and B (=W) tiles are K-contiguous row-major.
// Block: 256 threads (8 waves), 64x64 output tile, BK=64.
// Wave w owns row band tm=(w&3) and col tiles tn0=(w>>2)*2, tn0+1:
//   per K-step the A fragment is loaded once and reused by 2 WMMAs.
// 12 stages, 2 barriers/stage (24 total vs 48 at BK=32).
// -----------------------------------------------------------------------------
__global__ __launch_bounds__(256) void gemm_xwT_kernel(
    const float* __restrict__ X, const float* __restrict__ W,
    const float* __restrict__ bias, float* __restrict__ Y)
{
    __shared__ unsigned short sA[64 * 64];   // A tile, bf16, row stride 64 (8 KB)
    __shared__ unsigned short sB[64 * 64];   // B tile (rows = N, K-contig), bf16 (8 KB)

    const int tid  = threadIdx.x;
    const int lane = tid & 31;
    const int wave = tid >> 5;
    const int m0   = blockIdx.x * 64;
    const int n0   = blockIdx.y * 64;
    const int rA   = lane & 15;                 // fragment row within 16
    const int hbA  = (lane & 16) ? 8  : 0;      // A: segs at hbA, hbA+16
    const int hbB  = (lane & 16) ? 8  : 0;      // B: segs at hbB, hbB+16
    const int mo   = (lane & 16) ? 8  : 0;      // C/D row offset per lane half

    const int tm  = wave & 3;                   // row band
    const int tn0 = (wave >> 2) * 2;            // first of two col tiles

    f32x8 acc0 = {0.f,0.f,0.f,0.f,0.f,0.f,0.f,0.f};
    f32x8 acc1 = {0.f,0.f,0.f,0.f,0.f,0.f,0.f,0.f};

    const int lrow = tid >> 2;            // 0..63
    const int lcol = (tid & 3) * 16;      // 0,16,32,48

    const float* xbase = X + (size_t)(m0 + lrow) * BD_ + lcol;
    const float* wbase = W + (size_t)(n0 + lrow) * BD_ + lcol;

    for (int kt = 0; kt < BD_; kt += 64) {
        // prefetch next stage into L2 while we convert/compute this one
        if (kt + 64 < BD_) {
            __builtin_prefetch(xbase + kt + 64, 0, 3);   // -> global_prefetch_b8
            __builtin_prefetch(wbase + kt + 64, 0, 3);
        }
        const float4* xg = (const float4*)(xbase + kt);
        const float4* wg = (const float4*)(wbase + kt);
        unsigned short* da = &sA[lrow * 64 + lcol];
        unsigned short* db = &sB[lrow * 64 + lcol];
        for (int v = 0; v < 4; ++v) cvt4_store(da + v * 4, xg[v]);
        for (int v = 0; v < 4; ++v) cvt4_store(db + v * 4, wg[v]);
        __syncthreads();

        const unsigned short* arow  = &sA[(tm * 16 + rA) * 64];
        const unsigned short* brow0 = &sB[((tn0    ) * 16 + rA) * 64];
        const unsigned short* brow1 = &sB[((tn0 + 1) * 16 + rA) * 64];
        for (int kk = 0; kk < 64; kk += 32) {
            bf16x16 af = frag_ld(arow, kk + hbA, kk + hbA + 16);
            bf16x16 b0 = frag_ld(brow0, kk + hbB, kk + hbB + 16);
            acc0 = wmma_bf16(af, b0, acc0);
            bf16x16 b1 = frag_ld(brow1, kk + hbB, kk + hbB + 16);
            acc1 = wmma_bf16(af, b1, acc1);
        }
        __syncthreads();
    }

    {
        int col = n0 + tn0 * 16 + rA;
        float bv = bias ? bias[col] : 0.0f;
        for (int i = 0; i < 8; ++i)
            Y[(size_t)(m0 + tm * 16 + mo + i) * AD_ + col] = acc0[i] + bv;
    }
    {
        int col = n0 + (tn0 + 1) * 16 + rA;
        float bv = bias ? bias[col] : 0.0f;
        for (int i = 0; i < 8; ++i)
            Y[(size_t)(m0 + tm * 16 + mo + i) * AD_ + col] = acc1[i] + bv;
    }
}

// -----------------------------------------------------------------------------
// Fused attention for one (64-query block, head, batch):
//   scores = tanh(Q Kt) / sqrt(768); P = softmax_k(scores); ctx = P V
// Keeps the full 64x512 score panel in LDS (needs CDNA5's 320KB WGP LDS).
// grid = (S/64, H*B); block = 256 (8 waves). Dynamic LDS = 272 KB.
// -----------------------------------------------------------------------------
__global__ __launch_bounds__(256) void attn_kernel(
    const float* __restrict__ Q, const float* __restrict__ K,
    const float* __restrict__ V,
    float* __restrict__ ctxOut,   // ctx destination
    float* __restrict__ ctxOut2,  // optional second copy (final round), else null
    float* __restrict__ scOut)    // optional [H,B,S,S] probs (final round), else null
{
    extern __shared__ unsigned char smem[];
    unsigned short* qs = (unsigned short*)smem;            // 64x64 bf16  (8 KB)
    unsigned short* ks = qs + 64 * 64;                     // 64x64 bf16  (8 KB)
    float*          sc = (float*)(ks + 64 * 64);           // 64x512 f32  (128 KB)
    unsigned short* pb = (unsigned short*)(sc + 64 * 512); // 64x512 bf16 (64 KB)
    unsigned short* vt = pb + 64 * 512;                    // 64(d)x512(k) bf16 (64 KB)

    const int tid  = threadIdx.x;
    const int lane = tid & 31;
    const int wave = tid >> 5;
    const int hb   = blockIdx.y;           // = h*B + b
    const int h    = hb / B_;
    const int bat  = hb % B_;
    const int q0   = blockIdx.x * 64;
    const int rA   = lane & 15;
    const int hbA  = (lane & 16) ? 8  : 0;
    const int hbB  = (lane & 16) ? 8  : 0;
    const int mo   = (lane & 16) ? 8  : 0;

    const int lrow = tid >> 2;             // 0..63
    const int lc0  = (tid & 3) * 16;       // 0,16,32,48

    // ---- load Q block (64x64) -> qs ----
    {
        const float4* g = (const float4*)(Q + (size_t)(bat * S_ + q0 + lrow) * AD_ + h * DH_ + lc0);
        unsigned short* d = &qs[lrow * 64 + lc0];
        for (int v = 0; v < 4; ++v) cvt4_store(d + v * 4, g[v]);
    }
    // ---- load V transposed: vt[d][k] (coalesced over d) ----
    {
        int d  = tid & 63;
        int kb = tid >> 6;                 // 0..3
        for (int j = 0; j < 128; ++j) {
            int k = kb + j * 4;            // covers 0..511
            vt[d * 512 + k] = cvt_bf16(V[(size_t)(bat * S_ + k) * AD_ + h * DH_ + d]);
        }
    }
    __syncthreads();

    // ---- scores: loop 64-key chunks, WMMA Q*K^T, tanh/scale epilogue ----
    const float* kbase = K + (size_t)(bat * S_ + lrow) * AD_ + h * DH_ + lc0;
    for (int c = 0; c < 8; ++c) {
        if (c + 1 < 8)
            __builtin_prefetch(kbase + (size_t)(c + 1) * 64 * AD_, 0, 3);  // next K chunk
        {
            const float4* g = (const float4*)(kbase + (size_t)c * 64 * AD_);
            unsigned short* d = &ks[lrow * 64 + lc0];
            for (int v = 0; v < 4; ++v) cvt4_store(d + v * 4, g[v]);
        }
        __syncthreads();
        for (int tt = 0; tt < 2; ++tt) {
            int t = wave + tt * 8;
            int tm = t >> 2, tn = t & 3;
            f32x8 acc = {0.f,0.f,0.f,0.f,0.f,0.f,0.f,0.f};
            for (int kk = 0; kk < 64; kk += 32) {
                bf16x16 af = frag_ld(&qs[(tm * 16 + rA) * 64], kk + hbA, kk + hbA + 16);
                bf16x16 bfv = frag_ld(&ks[(tn * 16 + rA) * 64], kk + hbB, kk + hbB + 16);
                acc = wmma_bf16(af, bfv, acc);
            }
            int n = tn * 16 + rA;
            for (int i = 0; i < 8; ++i)
                sc[(tm * 16 + mo + i) * 512 + c * 64 + n] = tanhf(acc[i]) * INV_SCALE;
        }
        __syncthreads();
    }

    // ---- row softmax (wave32: one wave owns 8 rows) ----
    for (int i = 0; i < 8; ++i) {
        int r = wave * 8 + i;
        float* row = &sc[r * 512];
        float m = -1e30f;
        for (int j = lane; j < 512; j += 32) m = fmaxf(m, row[j]);
        for (int o = 16; o > 0; o >>= 1) m = fmaxf(m, __shfl_xor(m, o, 32));
        float s = 0.0f;
        for (int j = lane; j < 512; j += 32) {
            float e = __expf(row[j] - m);
            row[j] = e;
            s += e;
        }
        for (int o = 16; o > 0; o >>= 1) s += __shfl_xor(s, o, 32);
        float inv = 1.0f / s;
        for (int j = lane; j < 512; j += 32) {
            float p = row[j] * inv;
            pb[r * 512 + j] = cvt_bf16(p);
            if (scOut)
                scOut[((size_t)hb * S_ + (q0 + r)) * S_ + j] = p;
        }
    }
    __syncthreads();

    // ---- ctx = P @ V  (K = 512) ----
    for (int tt = 0; tt < 2; ++tt) {
        int t = wave + tt * 8;
        int tm = t >> 2, tn = t & 3;
        f32x8 acc = {0.f,0.f,0.f,0.f,0.f,0.f,0.f,0.f};
        for (int kk = 0; kk < 512; kk += 32) {
            bf16x16 af = frag_ld(&pb[(tm * 16 + rA) * 512], kk + hbA, kk + hbA + 16);
            bf16x16 bfv = frag_ld(&vt[(tn * 16 + rA) * 512], kk + hbB, kk + hbB + 16);
            acc = wmma_bf16(af, bfv, acc);
        }
        int n = tn * 16 + rA;
        for (int i = 0; i < 8; ++i) {
            size_t idx = (size_t)(bat * S_ + q0 + tm * 16 + mo + i) * AD_ + h * DH_ + n;
            ctxOut[idx] = acc[i];
            if (ctxOut2) ctxOut2[idx] = acc[i];
        }
    }
}

// -----------------------------------------------------------------------------
// Host launcher: 3 rounds of (3 GEMMs + fused attention).
// Workspace: ctx | Q | K | V, each 8192*768 f32 (~25 MB; 96 MB total).
// -----------------------------------------------------------------------------
extern "C" void kernel_launch(void* const* d_in, const int* in_sizes, int n_in,
                              void* d_out, int out_size, void* d_ws, size_t ws_size,
                              hipStream_t stream)
{
    (void)in_sizes; (void)n_in; (void)out_size; (void)ws_size;

    const float* s1  = (const float*)d_in[0];
    const float* s2  = (const float*)d_in[1];
    const float* Wq  = (const float*)d_in[2];
    const float* Wk  = (const float*)d_in[3];
    const float* Wv  = (const float*)d_in[4];
    const float* Wq1 = (const float*)d_in[5];
    const float* bq1 = (const float*)d_in[6];
    const float* Wk1 = (const float*)d_in[7];
    const float* bk1 = (const float*)d_in[8];
    const float* Wv1 = (const float*)d_in[9];
    const float* bv1 = (const float*)d_in[10];

    const size_t MS = (size_t)MROWS * AD_;   // 6,291,456 floats
    float* ctx = (float*)d_ws;
    float* Qb  = ctx + MS;
    float* Kb  = Qb + MS;
    float* Vb  = Kb + MS;

    float* out   = (float*)d_out;
    float* outC0 = out;             // ctx copy 1
    float* outC1 = out + MS;        // ctx copy 2
    float* outSc = out + 2 * MS;    // scores [H,B,S,S]

    dim3 blk(256);
    dim3 gG(MROWS / 64, AD_ / 64);  // 128 x 12
    dim3 gA(S_ / 64, H_ * B_);      // 8 x 192

    const size_t smem = (size_t)64*64*2 * 2      // qs + ks
                      + (size_t)64*512*4         // sc (f32)
                      + (size_t)64*512*2 * 2;    // pb + vt  => 278528 B
    (void)hipFuncSetAttribute((const void*)attn_kernel,
                              hipFuncAttributeMaxDynamicSharedMemorySize, (int)smem);

    // round 0 (no bias; s1 -> Q, s2 -> K,V)
    gemm_xwT_kernel<<<gG, blk, 0, stream>>>(s1, Wq, nullptr, Qb);
    gemm_xwT_kernel<<<gG, blk, 0, stream>>>(s2, Wk, nullptr, Kb);
    gemm_xwT_kernel<<<gG, blk, 0, stream>>>(s2, Wv, nullptr, Vb);
    attn_kernel<<<gA, blk, smem, stream>>>(Qb, Kb, Vb, ctx, nullptr, nullptr);

    // rounds 1..2 (bias; s1 = s2 = ctx)
    for (int j = 0; j < 2; ++j) {
        const float* Wqj = Wq1 + (size_t)j * AD_ * AD_;
        const float* Wkj = Wk1 + (size_t)j * AD_ * AD_;
        const float* Wvj = Wv1 + (size_t)j * AD_ * AD_;
        const float* bqj = bq1 + (size_t)j * AD_;
        const float* bkj = bk1 + (size_t)j * AD_;
        const float* bvj = bv1 + (size_t)j * AD_;
        const bool last = (j == 1);

        gemm_xwT_kernel<<<gG, blk, 0, stream>>>(ctx, Wqj, bqj, Qb);
        gemm_xwT_kernel<<<gG, blk, 0, stream>>>(ctx, Wkj, bkj, Kb);
        gemm_xwT_kernel<<<gG, blk, 0, stream>>>(ctx, Wvj, bvj, Vb);
        attn_kernel<<<gA, blk, smem, stream>>>(Qb, Kb, Vb,
                                               last ? outC0 : ctx,
                                               last ? outC1 : nullptr,
                                               last ? outSc : nullptr);
    }
}
